// DCGN_85873576117135
// MI455X (gfx1250) — matrix operand order repaired and checked by reading, not compiled, not run on hardware
//
#include <hip/hip_runtime.h>
#include <hip/hip_bf16.h>
#include <math.h>

// ---------------------------------------------------------------------------
// Problem constants (reference: N=5400, F=256, S=270, SHOT=20)
// ---------------------------------------------------------------------------
#define NN   5400
#define FF   256
#define SS   270
#define SHOT 20

typedef __bf16 bf16;
typedef bf16  v16bf __attribute__((ext_vector_type(16)));
typedef bf16  v8bf  __attribute__((ext_vector_type(8)));
typedef float v8f   __attribute__((ext_vector_type(8)));

// ---------------------------------------------------------------------------
// CDNA5 async copy: global -> LDS, 16 bytes per lane, tracked by ASYNCcnt.
// ---------------------------------------------------------------------------
__device__ __forceinline__ void async16(const bf16* g, const bf16* l) {
  unsigned lds = (unsigned)(unsigned long long)l;  // low 32 bits = LDS offset
  unsigned long long ga = (unsigned long long)g;
  asm volatile("global_load_async_to_lds_b128 %0, %1, off"
               :: "v"(lds), "v"(ga) : "memory");
}
__device__ __forceinline__ void wait_async0() {
  asm volatile("s_wait_asynccnt 0" ::: "memory");
}

// ---------------------------------------------------------------------------
// Tiled WMMA GEMM:  C(MxN) = A(MxKp) @ Bt(NxKp)^T, bf16 operands (zero-padded
// to Mp/Np/Kp tile multiples), f32 accumulate via v_wmma_f32_16x16x32_bf16.
// 256 threads (8 waves), 64x64 tile, K-step 32, double-buffered LDS filled by
// GLOBAL_LOAD_ASYNC_TO_LDS_B128 overlapped with the WMMAs of the prior chunk.
// ---------------------------------------------------------------------------
#define BM 64
#define BN 64
#define BK 32

template <int ELU>
__global__ __launch_bounds__(256) void wmma_gemm(const bf16* __restrict__ A,
                                                 const bf16* __restrict__ Bt,
                                                 float* __restrict__ C,
                                                 int M, int N, int Kp) {
  __shared__ bf16 Al[2][BM][BK];
  __shared__ bf16 Bl[2][BN][BK];

  const int tid  = threadIdx.x;
  const int lane = tid & 31;
  const int wave = tid >> 5;
  const int bm   = blockIdx.y * BM;
  const int bn   = blockIdx.x * BN;

  const int tm  = wave & 3;         // 16-row sub-tile
  const int tn0 = (wave >> 2) * 2;  // two 16-col sub-tiles
  const int h   = lane >> 4;
  const int l16 = lane & 15;

  // Staging: one 16B chunk per thread covers a 64x32 bf16 tile exactly.
  const int sm = tid >> 2;         // row 0..63
  const int sc = (tid & 3) * 8;    // element col 0/8/16/24
  const bf16* gA = A  + (size_t)(bm + sm) * Kp + sc;
  const bf16* gB = Bt + (size_t)(bn + sm) * Kp + sc;

  v8f acc0 = {}, acc1 = {};

  const int nch = Kp / BK;
  async16(gA, &Al[0][sm][sc]);
  async16(gB, &Bl[0][sm][sc]);
  wait_async0();
  __syncthreads();

  for (int ch = 0; ch < nch; ++ch) {
    const int cur = ch & 1;
    if (ch + 1 < nch) {  // prefetch next chunk while we do matrix math
      const int nxt = cur ^ 1;
      async16(gA + (ch + 1) * BK, &Al[nxt][sm][sc]);
      async16(gB + (ch + 1) * BK, &Bl[nxt][sm][sc]);
    }

    // A fragment (16x32): lane m=l16; element e -> k = 8h + (e&7) + 16*(e>>3)
    const int mrow = tm * 16 + l16;
    v8bf alo = *(const v8bf*)&Al[cur][mrow][8 * h];
    v8bf ahi = *(const v8bf*)&Al[cur][mrow][8 * h + 16];
    v16bf av;
#pragma unroll
    for (int e = 0; e < 8; ++e) { av[e] = alo[e]; av[e + 8] = ahi[e]; }

#pragma unroll
    for (int t = 0; t < 2; ++t) {
      // B fragment (32x16): lane n=l16; element e -> k = 8h + (e&7) + 16*(e>>3)
      const int ncol = (tn0 + t) * 16 + l16;
      v8bf blo = *(const v8bf*)&Bl[cur][ncol][8 * h];
      v8bf bhi = *(const v8bf*)&Bl[cur][ncol][8 * h + 16];
      v16bf bv;
#pragma unroll
      for (int e = 0; e < 8; ++e) { bv[e] = blo[e]; bv[e + 8] = bhi[e]; }
      if (t == 0)
        acc0 = __builtin_amdgcn_wmma_f32_16x16x32_bf16(false, av, false, bv,
                                                       (short)0, acc0, false, false);
      else
        acc1 = __builtin_amdgcn_wmma_f32_16x16x32_bf16(false, av, false, bv,
                                                       (short)0, acc1, false, false);
    }

    wait_async0();     // drain this wave's async copies into the next buffer
    __syncthreads();   // all waves' copies visible before anyone reads it
  }

  // Epilogue: c[v] is row (tm*16 + v + 8h), col (tn*16 + l16); fused ELU.
#pragma unroll
  for (int t = 0; t < 2; ++t) {
    v8f acc = t ? acc1 : acc0;
    int ncol = bn + (tn0 + t) * 16 + l16;
    if (ncol < N) {
#pragma unroll
      for (int v = 0; v < 8; ++v) {
        int row = bm + tm * 16 + v + 8 * h;
        if (row < M) {
          float x = acc[v];
          if (ELU) x = (x > 0.f) ? x : (expf(x) - 1.f);
          C[(size_t)row * N + ncol] = x;
        }
      }
    }
  }
}

// ---------------------------------------------------------------------------
// f32 -> zero-padded bf16 operand builders (row-major and transposed)
// ---------------------------------------------------------------------------
__global__ void cvt_bf16_kernel(const float* __restrict__ src, bf16* __restrict__ dst,
                                int M, int K, int Kp, int total) {
  int idx = blockIdx.x * blockDim.x + threadIdx.x;
  if (idx >= total) return;
  int m = idx / Kp, k = idx % Kp;
  float v = (m < M && k < K) ? src[(size_t)m * K + k] : 0.f;
  dst[idx] = (bf16)v;
}

__global__ void cvt_bf16_t_kernel(const float* __restrict__ src, bf16* __restrict__ dst,
                                  int K, int N, int Kp, int total) {
  int idx = blockIdx.x * blockDim.x + threadIdx.x;
  if (idx >= total) return;
  int n = idx / Kp, k = idx % Kp;
  float v = (k < K && n < N) ? src[(size_t)k * N + n] : 0.f;
  dst[idx] = (bf16)v;
}

// ---------------------------------------------------------------------------
// Layer-1 structured adjacency (exact segment algebra)
// ---------------------------------------------------------------------------
__global__ void seg_mean_kernel(const float* __restrict__ x,
                                const float* __restrict__ wnode,
                                float* __restrict__ Gm) {
  int s = blockIdx.x, f = threadIdx.x;
  float w = *wnode, acc = 0.f;
  for (int r = 0; r < SHOT; ++r) acc += x[(size_t)(s * SHOT + r) * FF + f];
  Gm[s * FF + f] = acc * w * (1.f / (float)SHOT);
}

__global__ void rownorm_kernel(const float* __restrict__ Gm,
                               float* __restrict__ nrmS) {
  __shared__ float red[FF];
  int s = blockIdx.x, f = threadIdx.x;
  float v = Gm[s * FF + f];
  red[f] = v * v;
  __syncthreads();
  for (int st = FF / 2; st > 0; st >>= 1) {
    if (f < st) red[f] += red[f + st];
    __syncthreads();
  }
  if (f == 0) nrmS[s] = sqrtf(red[0]);
}

__global__ void gram_cos_kernel(const float* __restrict__ Gm,
                                const float* __restrict__ nrmS,
                                float* __restrict__ Cs) {
  int idx = blockIdx.x * blockDim.x + threadIdx.x;
  if (idx >= SS * SS) return;
  int i = idx / SS, j = idx % SS;
  float acc = 0.f;
  for (int f = 0; f < FF; ++f) acc += Gm[i * FF + f] * Gm[j * FF + f];
  Cs[idx] = acc / (nrmS[i] * nrmS[j]);
}

__global__ void dinv1_kernel(const float* __restrict__ Cs,
                             float* __restrict__ dinv1) {
  int j = blockIdx.x * blockDim.x + threadIdx.x;
  if (j >= NN) return;
  int sj = j / SHOT;
  int t = NN - j;
  int st = t / SHOT, p = t % SHOT;
  float acc = 0.f;
  for (int sp = 0; sp < st; ++sp) acc += Cs[sp * SS + sj];
  float colsum = (float)SHOT * acc + ((st < SS) ? (float)p * Cs[st * SS + sj] : 0.f)
                 + (float)j + 1.f;
  dinv1[j] = rsqrtf(colsum);
}

__global__ void zseg_kernel(const float* __restrict__ x,
                            const float* __restrict__ wnode,
                            const float* __restrict__ dinv1,
                            float* __restrict__ Zseg) {
  int s = blockIdx.x, f = threadIdx.x;
  float w = *wnode, acc = 0.f;
  for (int r = 0; r < SHOT; ++r) {
    int i = s * SHOT + r;
    acc += dinv1[i] * x[(size_t)i * FF + f];
  }
  Zseg[s * FF + f] = acc * w;
}

__global__ void zprefix_kernel(const float* __restrict__ Zseg,
                               float* __restrict__ ZsegPre) {
  int f = threadIdx.x;
  float acc = 0.f;
  ZsegPre[f] = 0.f;
  for (int s = 0; s < SS; ++s) {
    acc += Zseg[s * FF + f];
    ZsegPre[(s + 1) * FF + f] = acc;
  }
}

__global__ void d1_kernel(const float* __restrict__ Cs,
                          const float* __restrict__ Zseg,
                          float* __restrict__ D1) {
  int k = blockIdx.x, f = threadIdx.x;
  int lim = SS - 1 - k;
  float acc = 0.f;
  for (int sj = 0; sj < lim; ++sj) acc += Cs[k * SS + sj] * Zseg[sj * FF + f];
  D1[k * FF + f] = acc;
}

__global__ void y1_kernel(const float* __restrict__ x,
                          const float* __restrict__ wnode,
                          const float* __restrict__ Cs,
                          const float* __restrict__ dinv1,
                          const float* __restrict__ Zseg,
                          const float* __restrict__ ZsegPre,
                          const float* __restrict__ D1,
                          float* __restrict__ Y1) {
  int i = blockIdx.x, f = threadIdx.x;
  int k = i / SHOT, r = i % SHOT;
  float w = *wnode;
  int sE = SS - 1 - k;
  float csE = Cs[k * SS + sE];
  float S1, PzT;
  if (r == 0) {
    S1 = D1[k * FF + f] + csE * Zseg[sE * FF + f];
    PzT = ZsegPre[(sE + 1) * FF + f];
  } else {
    int pt = SHOT - r;
    int base = sE * SHOT;
    float Pp = 0.f;
    for (int q = 0; q < pt; ++q) {
      int jj = base + q;
      Pp += dinv1[jj] * x[(size_t)jj * FF + f];
    }
    Pp *= w;
    S1 = D1[k * FF + f] + csE * Pp;
    PzT = ZsegPre[sE * FF + f] + Pp;
  }
  float Ztot = ZsegPre[SS * FF + f];
  float di = dinv1[i];
  Y1[(size_t)i * FF + f] = di * (S1 + Ztot - PzT + di * w * x[(size_t)i * FF + f]);
}

// ---------------------------------------------------------------------------
// Attention pooling + node conv (groups of 3)
// ---------------------------------------------------------------------------
template <int F>
__global__ void attpool_kernel(const float* __restrict__ H,
                               const float* __restrict__ apw,
                               const float* __restrict__ apb,
                               float* __restrict__ rep) {
  __shared__ float red[6][F];
  int g = blockIdx.x, f = threadIdx.x;
#pragma unroll
  for (int k = 0; k < 3; ++k) {
    float v = H[(size_t)(3 * g + k) * F + f];
    red[k][f] = v * apw[f];
    red[3 + k][f] = v;
  }
  __syncthreads();
  for (int st = F / 2; st > 0; st >>= 1) {
    if (f < st)
      for (int k = 0; k < 6; ++k) red[k][f] += red[k][f + st];
    __syncthreads();
  }
  if (f < 3) {
    float b = *apb;
    float d0 = red[0][0] + b, d1 = red[1][0] + b, d2 = red[2][0] + b;
    float m = fmaxf(d0, fmaxf(d1, d2));
    float e0 = expf(d0 - m), e1 = expf(d1 - m), e2 = expf(d2 - m);
    float inv = 1.f / (e0 + e1 + e2);
    float e = (f == 0) ? e0 : ((f == 1) ? e1 : e2);
    rep[g * 3 + f] = e * inv * red[3 + f][0];
  }
}

__global__ void nodeconv_kernel(const float* __restrict__ H,
                                const float* __restrict__ wnc,
                                float* __restrict__ xc, int G, int F) {
  int idx = blockIdx.x * blockDim.x + threadIdx.x;
  if (idx >= G * F) return;
  int g = idx / F, f = idx % F;
  float acc = 0.f;
#pragma unroll
  for (int k = 0; k < 3; ++k) acc += wnc[k] * H[(size_t)(3 * g + k) * F + f];
  xc[idx] = acc;
}

// ---------------------------------------------------------------------------
// Dense (3-dim rep) adjacency for layers 2/3
// ---------------------------------------------------------------------------
__global__ void norm3_kernel(const float* __restrict__ rep,
                             float* __restrict__ nr, int Mr) {
  int i = blockIdx.x * blockDim.x + threadIdx.x;
  if (i >= Mr) return;
  float a = rep[3 * i], b = rep[3 * i + 1], c = rep[3 * i + 2];
  nr[i] = sqrtf(a * a + b * b + c * c);
}

__global__ void colsumR_kernel(const float* __restrict__ rep,
                               const float* __restrict__ nr,
                               float* __restrict__ dinvR, int Mr) {
  int j = blockIdx.x * blockDim.x + threadIdx.x;
  if (j >= Mr) return;
  float rj0 = rep[3 * j], rj1 = rep[3 * j + 1], rj2 = rep[3 * j + 2];
  float invnj = 1.f / nr[j];
  int t = Mr - j;
  float acc = 0.f;
  for (int i = 0; i < t; ++i) {
    float d = rep[3 * i] * rj0 + rep[3 * i + 1] * rj1 + rep[3 * i + 2] * rj2;
    acc += d / nr[i];
  }
  dinvR[j] = rsqrtf(acc * invnj + (float)j + 1.f);
}

__global__ void adjmat_kernel(const float* __restrict__ rep,
                              const float* __restrict__ nr,
                              const float* __restrict__ dinvR,
                              float* __restrict__ adj, int Mr) {
  int idx = blockIdx.x * blockDim.x + threadIdx.x;
  if (idx >= Mr * Mr) return;
  int i = idx / Mr, j = idx % Mr;
  float v;
  if (i + j < Mr) {
    float d = rep[3 * i] * rep[3 * j] + rep[3 * i + 1] * rep[3 * j + 1] +
              rep[3 * i + 2] * rep[3 * j + 2];
    v = d / (nr[i] * nr[j]);
  } else {
    v = 1.f;
  }
  if (i == j) v += 1.f;
  adj[idx] = dinvR[i] * dinvR[j] * v;
}

// ---------------------------------------------------------------------------
// Host-side orchestration
// ---------------------------------------------------------------------------
static inline int ceil_div(int a, int b) { return (a + b - 1) / b; }

extern "C" void kernel_launch(void* const* d_in, const int* in_sizes, int n_in,
                              void* d_out, int out_size, void* d_ws, size_t ws_size,
                              hipStream_t stream) {
  (void)in_sizes; (void)n_in; (void)out_size; (void)ws_size;

  const float* x     = (const float*)d_in[0];
  const float* wnode = (const float*)d_in[2];
  const float* P1    = (const float*)d_in[3];
  const float* wnc1  = (const float*)d_in[4];
  const float* ap1w  = (const float*)d_in[5];
  const float* ap1b  = (const float*)d_in[6];
  const float* P2    = (const float*)d_in[7];
  const float* wnc2  = (const float*)d_in[8];
  const float* ap2w  = (const float*)d_in[9];
  const float* ap2b  = (const float*)d_in[10];
  const float* P3    = (const float*)d_in[11];
  float* out = (float*)d_out;

  // Byte-granular workspace carving (256B aligned slices).
  char* base = (char*)d_ws;
  size_t o = 0;
  auto allocB = [&](size_t bytes) {
    void* p = base + o;
    o = (o + bytes + 255) & ~(size_t)255;
    return p;
  };
  auto allocF = [&](size_t n) { return (float*)allocB(n * sizeof(float)); };
  auto allocH = [&](size_t n) { return (bf16*)allocB(n * sizeof(bf16)); };

  float* Gm      = allocF((size_t)SS * FF);
  float* nrmS    = allocF(SS);
  float* Cs      = allocF((size_t)SS * SS);
  float* dinv1   = allocF(NN);
  float* Zseg    = allocF((size_t)SS * FF);
  float* ZsegPre = allocF((size_t)(SS + 1) * FF);
  float* D1      = allocF((size_t)SS * FF);
  float* Y1      = allocF((size_t)NN * FF);
  float* H1      = allocF((size_t)NN * FF);
  float* rep1    = allocF(1800 * 3);
  float* xc1     = allocF((size_t)1800 * 256);
  float* nr1     = allocF(1800);
  float* dR1     = allocF(1800);
  float* adj2    = allocF((size_t)1800 * 1800);
  float* Y2      = allocF((size_t)1800 * 256);
  float* H2      = allocF((size_t)1800 * 64);
  float* rep2    = allocF(600 * 3);
  float* xc2     = allocF((size_t)600 * 64);
  float* nr2     = allocF(600);
  float* dR2     = allocF(600);
  float* adj3    = allocF((size_t)600 * 600);
  float* Y3      = allocF((size_t)600 * 64);

  // Zero-padded bf16 GEMM operands (pads: M->64, N->64, K->32 multiples).
  bf16* Y1bf   = allocH((size_t)5440 * 256);   // 5400x256
  bf16* P1t    = allocH((size_t)256 * 256);    // (256x256)^T
  bf16* adj2bf = allocH((size_t)1856 * 1824);  // 1800x1800 -> 1856x1824
  bf16* xc1t   = allocH((size_t)256 * 1824);   // (1800x256)^T
  bf16* Y2bf   = allocH((size_t)1856 * 256);   // 1800x256
  bf16* P2t    = allocH((size_t)64 * 256);     // (256x64)^T
  bf16* adj3bf = allocH((size_t)640 * 608);    // 600x600 -> 640x608
  bf16* xc2t   = allocH((size_t)64 * 608);     // (600x64)^T
  bf16* Y3bf   = allocH((size_t)640 * 64);     // 600x64
  bf16* P3t    = allocH((size_t)64 * 64);      // (64x32)^T padded

  const int T = 256;

  // ---- Layer 1: structured adjacency (exact segment algebra) ----
  seg_mean_kernel<<<SS, FF, 0, stream>>>(x, wnode, Gm);
  rownorm_kernel<<<SS, FF, 0, stream>>>(Gm, nrmS);
  gram_cos_kernel<<<ceil_div(SS * SS, T), T, 0, stream>>>(Gm, nrmS, Cs);
  dinv1_kernel<<<ceil_div(NN, T), T, 0, stream>>>(Cs, dinv1);
  zseg_kernel<<<SS, FF, 0, stream>>>(x, wnode, dinv1, Zseg);
  zprefix_kernel<<<1, FF, 0, stream>>>(Zseg, ZsegPre);
  d1_kernel<<<SS, FF, 0, stream>>>(Cs, Zseg, D1);
  y1_kernel<<<NN, FF, 0, stream>>>(x, wnode, Cs, dinv1, Zseg, ZsegPre, D1, Y1);

  // H1 = elu(Y1 @ P1) via WMMA
  cvt_bf16_kernel<<<ceil_div(5440 * 256, T), T, 0, stream>>>(Y1, Y1bf, 5400, 256, 256, 5440 * 256);
  cvt_bf16_t_kernel<<<ceil_div(256 * 256, T), T, 0, stream>>>(P1, P1t, 256, 256, 256, 256 * 256);
  {
    dim3 g(ceil_div(256, BN), ceil_div(NN, BM));
    wmma_gemm<1><<<g, 256, 0, stream>>>(Y1bf, P1t, H1, NN, 256, 256);
  }
  attpool_kernel<256><<<1800, 256, 0, stream>>>(H1, ap1w, ap1b, rep1);
  nodeconv_kernel<<<ceil_div(1800 * 256, T), T, 0, stream>>>(H1, wnc1, xc1, 1800, 256);

  // ---- Layer 2 ----
  norm3_kernel<<<ceil_div(1800, T), T, 0, stream>>>(rep1, nr1, 1800);
  colsumR_kernel<<<ceil_div(1800, T), T, 0, stream>>>(rep1, nr1, dR1, 1800);
  adjmat_kernel<<<ceil_div(1800 * 1800, T), T, 0, stream>>>(rep1, nr1, dR1, adj2, 1800);
  cvt_bf16_kernel<<<ceil_div(1856 * 1824, T), T, 0, stream>>>(adj2, adj2bf, 1800, 1800, 1824, 1856 * 1824);
  cvt_bf16_t_kernel<<<ceil_div(256 * 1824, T), T, 0, stream>>>(xc1, xc1t, 1800, 256, 1824, 256 * 1824);
  {
    dim3 g(ceil_div(256, BN), ceil_div(1800, BM));
    wmma_gemm<0><<<g, 256, 0, stream>>>(adj2bf, xc1t, Y2, 1800, 256, 1824);
  }
  cvt_bf16_kernel<<<ceil_div(1856 * 256, T), T, 0, stream>>>(Y2, Y2bf, 1800, 256, 256, 1856 * 256);
  cvt_bf16_t_kernel<<<ceil_div(64 * 256, T), T, 0, stream>>>(P2, P2t, 256, 64, 256, 64 * 256);
  {
    dim3 g(ceil_div(64, BN), ceil_div(1800, BM));
    wmma_gemm<1><<<g, 256, 0, stream>>>(Y2bf, P2t, H2, 1800, 64, 256);
  }
  attpool_kernel<64><<<600, 64, 0, stream>>>(H2, ap2w, ap2b, rep2);
  nodeconv_kernel<<<ceil_div(600 * 64, T), T, 0, stream>>>(H2, wnc2, xc2, 600, 64);

  // ---- Layer 3 ----
  norm3_kernel<<<ceil_div(600, T), T, 0, stream>>>(rep2, nr2, 600);
  colsumR_kernel<<<ceil_div(600, T), T, 0, stream>>>(rep2, nr2, dR2, 600);
  adjmat_kernel<<<ceil_div(600 * 600, T), T, 0, stream>>>(rep2, nr2, dR2, adj3, 600);
  cvt_bf16_kernel<<<ceil_div(640 * 608, T), T, 0, stream>>>(adj3, adj3bf, 600, 600, 608, 640 * 608);
  cvt_bf16_t_kernel<<<ceil_div(64 * 608, T), T, 0, stream>>>(xc2, xc2t, 600, 64, 608, 64 * 608);
  {
    dim3 g(ceil_div(64, BN), ceil_div(600, BM));
    wmma_gemm<0><<<g, 256, 0, stream>>>(adj3bf, xc2t, Y3, 600, 64, 608);
  }
  cvt_bf16_kernel<<<ceil_div(640 * 64, T), T, 0, stream>>>(Y3, Y3bf, 600, 64, 64, 640 * 64);
  cvt_bf16_t_kernel<<<ceil_div(64 * 64, T), T, 0, stream>>>(P3, P3t, 64, 32, 64, 64 * 64);
  {
    dim3 g(ceil_div(32, BN), ceil_div(600, BM));
    wmma_gemm<1><<<g, 256, 0, stream>>>(Y3bf, P3t, out, 600, 32, 64);
  }
}
